// EEGGCN_51900384805431
// MI455X (gfx1250) — matrix operand order repaired and checked by reading, not compiled, not run on hardware
//
#include <hip/hip_runtime.h>

typedef __attribute__((ext_vector_type(16))) __bf16 v16bf;
typedef __attribute__((ext_vector_type(8)))  float  v8f;

#define NN 50000
#define NE 800000
#define NG 256
#define CH 128
#define IN_C 256
#define NCLS 6
#define BN_EPS 1e-5f

// ---------------- degree / dinv ----------------
__global__ void fill_one(float* __restrict__ p, int n) {
    int i = blockIdx.x * blockDim.x + threadIdx.x;
    if (i < n) p[i] = 1.0f;   // self-loop contributes degree 1
}

__global__ void deg_scatter(const int* __restrict__ col, float* __restrict__ deg, int ne) {
    int e = blockIdx.x * blockDim.x + threadIdx.x;
    if (e < ne) atomicAdd(&deg[col[e]], 1.0f);
}

__global__ void deg_to_dinv(float* __restrict__ deg, int n) {
    int i = blockIdx.x * blockDim.x + threadIdx.x;
    if (i < n) deg[i] = rsqrtf(deg[i]);
}

// ---------------- bf16 WMMA GEMM: H[M,N] = X[M,K] @ W[K,N], N == CH ----------------
// Workgroup: 256 threads = 8 waves. W is staged once into LDS as bf16,
// TRANSPOSED to [col][k] so each lane's B fragment (16 contiguous K values)
// is one 32-byte LDS read. Each wave computes a 16(M) x 128(N) stripe with
// 8 f32 accumulators. Per K-chunk: load A once (global, f32->bf16), issue
// ALL 16 ds_load_b128 for the 8 B fragments into distinct registers, then
// run the 8 WMMAs — lets the scheduler overlap DS returns with the XDL pipe
// instead of a full s_wait_dscnt 0 drain before each matrix op.
__global__ void gemm_wmma_bf16(const float* __restrict__ X, const float* __restrict__ W,
                               float* __restrict__ H, int M, int K) {
    extern __shared__ __bf16 wlds[];   // [CH][K] : wlds[n*K + k] = bf16(W[k][n])

    // cooperative load + transpose + f32->bf16 convert of W (K x CH)
    int khalf = K >> 1;
    for (int idx = threadIdx.x; idx < khalf * CH; idx += blockDim.x) {
        int n  = idx / khalf;
        int k2 = idx - n * khalf;
        float w0 = W[(size_t)(2 * k2)     * CH + n];
        float w1 = W[(size_t)(2 * k2 + 1) * CH + n];
        wlds[(size_t)n * K + 2 * k2]     = (__bf16)w0;
        wlds[(size_t)n * K + 2 * k2 + 1] = (__bf16)w1;
    }
    __syncthreads();

    int tilesM = M >> 4;
    int tM = blockIdx.x * (blockDim.x >> 5) + (threadIdx.x >> 5);  // one wave = one 16-row stripe
    if (tM >= tilesM) return;                                      // wave-uniform exit (after barrier)

    int lane = threadIdx.x & 31;
    int hv   = lane >> 4;          // lane half selects K sub-ranges
    int l16  = lane & 15;

    const float* xrow = X + (size_t)(tM * 16 + l16) * K;   // A: row = M

    v8f acc[8] = {};
    for (int k0 = 0; k0 < K; k0 += 32) {
        // A 16x32 bf16 fragment: elem i -> K = ka + i (contiguous, matches B)
        v16bf a;
        int ka = k0 + hv * 16;
        #pragma unroll
        for (int i = 0; i < 8; ++i) {
            a[i]     = (__bf16)xrow[ka + i];
            a[i + 8] = (__bf16)xrow[ka + 8 + i];
        }

        // stage all 8 B fragments first (distinct registers -> overlapped DS)
        int kb = k0 + hv * 16;
        v16bf bfrag[8];
        #pragma unroll
        for (int tn = 0; tn < 8; ++tn) {
            // B 32x16 fragment: elem i -> K = kb + i, column tn*16 + l16 (contiguous in LDS)
            bfrag[tn] = *(const v16bf*)(wlds + (size_t)(tn * 16 + l16) * K + kb);
        }

        // 8 N-tiles share the A fragment
        #pragma unroll
        for (int tn = 0; tn < 8; ++tn) {
            acc[tn] = __builtin_amdgcn_wmma_f32_16x16x32_bf16(
                false, a, false, bfrag[tn], (short)0, acc[tn], false, false);
        }
    }

    // C/D layout: VGPR r -> (M = 8*half + r, N = l16)
    #pragma unroll
    for (int tn = 0; tn < 8; ++tn) {
        float* hrow = H + (size_t)(tM * 16 + hv * 8) * CH + (tn * 16 + l16);
        #pragma unroll
        for (int r = 0; r < 8; ++r) hrow[(size_t)r * CH] = acc[tn][r];
    }
}

// ---------------- self-loop init: A[n] = H[n] * dinv[n]^2 ----------------
__global__ void self_loop_init(const float* __restrict__ H, const float* __restrict__ dinv,
                               float* __restrict__ A, int n) {
    int t = blockIdx.x * blockDim.x + threadIdx.x;  // n * 32 threads (4 ch each)
    int node = t >> 5;
    if (node >= n) return;
    int ch = (t & 31) * 4;
    float d = dinv[node];
    float s = d * d;
    float4 h = *(const float4*)(H + (size_t)node * CH + ch);
    float4 o = make_float4(h.x * s, h.y * s, h.z * s, h.w * s);
    *(float4*)(A + (size_t)node * CH + ch) = o;
}

// ---------------- edge scatter: A[c] += H[r] * dinv[r]*dinv[c] ----------------
// One wave per edge; lane covers 4 channels (float4 gather, 4 f32 atomics).
// h (25.6 MB) is L2-resident (192 MB L2), so gathers/atomics stay on-chip.
__global__ void edge_scatter(const int* __restrict__ row, const int* __restrict__ col,
                             const float* __restrict__ H, const float* __restrict__ dinv,
                             float* __restrict__ A, int ne) {
    int e = blockIdx.x * (blockDim.x >> 5) + (threadIdx.x >> 5);
    if (e >= ne) return;
    int lane = threadIdx.x & 31;
    int r = row[e], c = col[e];
    float nrm = dinv[r] * dinv[c];
    int ch = lane * 4;
    float4 h = *(const float4*)(H + (size_t)r * CH + ch);
    float* dst = A + (size_t)c * CH + ch;
    atomicAdd(dst + 0, h.x * nrm);
    atomicAdd(dst + 1, h.y * nrm);
    atomicAdd(dst + 2, h.z * nrm);
    atomicAdd(dst + 3, h.w * nrm);
}

// ---------------- bias + BN + ReLU (in place) ----------------
__global__ void bias_bn_relu(float* __restrict__ A, const float* __restrict__ b,
                             const float* __restrict__ g, const float* __restrict__ bt,
                             const float* __restrict__ m, const float* __restrict__ v, int n) {
    int t = blockIdx.x * blockDim.x + threadIdx.x;
    if (t >= n * CH) return;
    int ch = t & (CH - 1);
    float x = A[t] + b[ch];
    x = (x - m[ch]) * rsqrtf(v[ch] + BN_EPS) * g[ch] + bt[ch];
    A[t] = fmaxf(x, 0.0f);
}

// ---------------- pooling ----------------
__global__ void zero_buf(float* __restrict__ p, int n) {
    int i = blockIdx.x * blockDim.x + threadIdx.x;
    if (i < n) p[i] = 0.0f;
}

__global__ void pool_acc(const float* __restrict__ A, const int* __restrict__ batch,
                         float* __restrict__ pooled, float* __restrict__ cnt, int n) {
    int t = blockIdx.x * blockDim.x + threadIdx.x;  // n * 32 threads
    int node = t >> 5;
    if (node >= n) return;
    int lane = t & 31;
    int g = batch[node];
    int ch = lane * 4;
    float4 h = *(const float4*)(A + (size_t)node * CH + ch);
    float* dst = pooled + (size_t)g * CH + ch;
    atomicAdd(dst + 0, h.x);
    atomicAdd(dst + 1, h.y);
    atomicAdd(dst + 2, h.z);
    atomicAdd(dst + 3, h.w);
    if (lane == 0) atomicAdd(&cnt[g], 1.0f);
}

// ---------------- final classifier: out[g,cls] = mean(pooled[g]) @ lin_W + lin_b ----------------
__global__ void final_linear(const float* __restrict__ pooled, const float* __restrict__ cnt,
                             const float* __restrict__ lw, const float* __restrict__ lb,
                             float* __restrict__ out) {
    int t = blockIdx.x * blockDim.x + threadIdx.x;
    if (t >= NG * NCLS) return;
    int g = t / NCLS;
    int cls = t - g * NCLS;
    float c = fmaxf(cnt[g], 1.0f);
    const float* pg = pooled + (size_t)g * CH;
    float s = 0.0f;
    #pragma unroll 8
    for (int k = 0; k < CH; ++k) s += pg[k] * lw[k * NCLS + cls];
    out[t] = s / c + lb[cls];
}

extern "C" void kernel_launch(void* const* d_in, const int* in_sizes, int n_in,
                              void* d_out, int out_size, void* d_ws, size_t ws_size,
                              hipStream_t stream) {
    (void)in_sizes; (void)n_in; (void)out_size; (void)ws_size;

    const float* x     = (const float*)d_in[0];
    const int*   ei    = (const int*)d_in[1];      // edge_index [2, NE]
    const int*   rowp  = ei;
    const int*   colp  = ei + NE;
    const int*   batch = (const int*)d_in[2];
    const float* lin_W = (const float*)d_in[21];
    const float* lin_b = (const float*)d_in[22];
    float*       out   = (float*)d_out;

    // workspace layout (floats)
    float* bufA   = (float*)d_ws;                       // GEMM output  [NN, CH]
    float* bufB   = bufA + (size_t)NN * CH;             // aggregated / activated [NN, CH]
    float* dinv   = bufB + (size_t)NN * CH;             // [NN] (holds deg, then rsqrt(deg))
    float* pooled = dinv + ((NN + 63) & ~63);           // [NG, CH]
    float* cnt    = pooled + (size_t)NG * CH;           // [NG]

    // degrees -> dinv
    fill_one   <<<(NN + 255) / 256, 256, 0, stream>>>(dinv, NN);
    deg_scatter<<<(NE + 255) / 256, 256, 0, stream>>>(colp, dinv, NE);
    deg_to_dinv<<<(NN + 255) / 256, 256, 0, stream>>>(dinv, NN);

    const float* in = x;
    int K = IN_C;
    for (int layer = 0; layer < 3; ++layer) {
        const float* W  = (const float*)d_in[3 + layer * 6 + 0];
        const float* b  = (const float*)d_in[3 + layer * 6 + 1];
        const float* g  = (const float*)d_in[3 + layer * 6 + 2];
        const float* bt = (const float*)d_in[3 + layer * 6 + 3];
        const float* m  = (const float*)d_in[3 + layer * 6 + 4];
        const float* v  = (const float*)d_in[3 + layer * 6 + 5];

        int tilesM = NN / 16;                           // 3125 16-row stripes (one per wave)
        int blocks = (tilesM + 7) / 8;                  // 8 waves / 256-thread block
        size_t lds = (size_t)K * CH * sizeof(__bf16);   // 64 KB (K=256) or 32 KB (K=128)
        gemm_wmma_bf16<<<blocks, 256, lds, stream>>>(in, W, bufA, NN, K);

        self_loop_init<<<(NN * 32 + 255) / 256, 256, 0, stream>>>(bufA, dinv, bufB, NN);
        edge_scatter  <<<(NE + 7) / 8, 256, 0, stream>>>(rowp, colp, bufA, dinv, bufB, NE);
        bias_bn_relu  <<<(NN * CH + 255) / 256, 256, 0, stream>>>(bufB, b, g, bt, m, v, NN);

        in = bufB;
        K = CH;
    }

    // global mean pool + classifier
    zero_buf<<<(NG * CH + NG + 255) / 256, 256, 0, stream>>>(pooled, NG * CH + NG);
    pool_acc<<<(NN * 32 + 255) / 256, 256, 0, stream>>>(bufB, batch, pooled, cnt, NN);
    final_linear<<<(NG * NCLS + 255) / 256, 256, 0, stream>>>(pooled, cnt, lin_W, lin_b, out);
}
